// Nonlinear_23106924052737
// MI455X (gfx1250) — matrix operand order repaired and checked by reading, not compiled, not run on hardware
//
#include <hip/hip_runtime.h>

// ---- problem constants (match reference) ----
#define NFFT   4194304
#define NMODES 2
#define NTAPS  101
#define NOUT   (NFFT - NTAPS + 1)   // 4,194,204
#define N0     (NTAPS / 2)          // 50
#define GAMMA  0.0016567f
#define DZ     80000.0f

// ---- tiling ----
#define BT      1024                // output times per workgroup (1 WMMA tile per wave)
#define KPRIME  112                 // padded time-offset range (k' = 0..111; used 0..107)
#define KP      (KPRIME * NMODES)   // padded K of the Toeplitz GEMM = 224 (7 x 32)
#define NCHUNK  (KP / 32)           // 7 WMMA k-steps per tile
#define PTIMES  (BT + KPRIME)       // input times resident in LDS = 1136
#define NTILES  (BT / 128)          // 8 WMMA tiles (each = 128 out times x 2 modes)

typedef __attribute__((ext_vector_type(16))) _Float16 v16h;
typedef __attribute__((ext_vector_type(8)))  _Float16 v8h;
typedef __attribute__((ext_vector_type(4)))  _Float16 v4h;
typedef __attribute__((ext_vector_type(8)))  float    v8f;

__global__ __launch_bounds__(256) void nl_phase_wmma_kernel(
    const float* __restrict__ xr,
    const float* __restrict__ xi,
    const float* __restrict__ W,     // [NTAPS][in=2][out=2]
    const float* __restrict__ bias,  // [2]
    const float* __restrict__ power, // [1]
    float*       __restrict__ out)   // [NOUT][2] complex interleaved (re,im)
{
    __shared__ __align__(16) _Float16 Plds[PTIMES * NMODES]; // f16 power, interleaved [t][mode]
    __shared__ __align__(16) _Float16 Blds[16 * KP];         // Toeplitz B, column-major [col][k]
    __shared__ __align__(16) float    XRl[PTIMES * NMODES];  // f32 x kept for final rotation
    __shared__ __align__(16) float    XIl[PTIMES * NMODES];

    const int tid = threadIdx.x;
    const int blk = blockIdx.x * BT;      // first output time of this block

    // dBm -> W per mode, fold all constants into one phase coefficient
    const float pw   = power[0];
    const float coef = GAMMA * DZ * (0.001f * __powf(10.0f, pw * 0.1f) / (float)NMODES);

    // ---- Phase 0: build Toeplitz weight matrix B[k'*2+i][j*2+o] = W[k'-j, i, o] ----
    for (int e = tid; e < 16 * KP; e += 256) {
        const int col = e / KP;            // col = j*2 + o
        const int k   = e - col * KP;      // k   = k'*2 + i
        const int j   = col >> 1, o = col & 1;
        const int kp  = k >> 1,  i = k & 1;
        const int tap = kp - j;
        float v = 0.0f;
        if (tap >= 0 && tap < NTAPS) v = W[tap * 4 + i * 2 + o];
        Blds[col * KP + k] = (_Float16)v;
    }

    // ---- Phase 1: P = |x|^2 -> f16 LDS; stash f32 x for the rotation (b128 loads) ----
    // Each chunk u covers 2 time samples (= 4 interleaved floats) of xr and xi.
    for (int u = tid; u < PTIMES / 2; u += 256) {
        const int gt = blk + 2 * u;        // even; gt < NFFT implies gt+1 < NFFT (NFFT even)
        float4 r = make_float4(0.f, 0.f, 0.f, 0.f);
        float4 m = make_float4(0.f, 0.f, 0.f, 0.f);
        if (gt < NFFT) {
            r = ((const float4*)xr)[blk / 2 + u];
            m = ((const float4*)xi)[blk / 2 + u];
        }
        v4h p;
        p[0] = (_Float16)(r.x * r.x + m.x * m.x);
        p[1] = (_Float16)(r.y * r.y + m.y * m.y);
        p[2] = (_Float16)(r.z * r.z + m.z * m.z);
        p[3] = (_Float16)(r.w * r.w + m.w * m.w);
        *(v4h*)(&Plds[u * 4]) = p;
        ((float4*)XRl)[u] = r;
        ((float4*)XIl)[u] = m;
    }

    __syncthreads();

    // ---- Phase 2: Toeplitz GEMM via v_wmma_f32_16x16x32_f16 (one tile per wave) ----
    const int wave  = tid >> 5;
    const int lane  = tid & 31;
    const int laneM = lane & 15;           // A row / B col / C col index
    const int hi    = lane >> 4;           // lane half (selects K sub-range / C row half)

    const float bmode = bias[laneM & 1];   // output-mode bias for this lane's column
    const int   jcol  = laneM >> 1;        // time offset j encoded in the column
    const int   omode = laneM & 1;

    const int tl = wave * 128;             // tile's first output time (block-local)

    const _Float16* ap = &Plds[(tl + laneM * 8) * 2 + (hi ? 8 : 0)];
    const _Float16* bp = &Blds[laneM * KP + (hi ? 8 : 0)];

    // Prefetch all fragments into registers so the DS->WMMA waits can cascade.
    v16h af[NCHUNK], bf[NCHUNK];
#pragma unroll
    for (int c = 0; c < NCHUNK; ++c) {
        v8h a0 = *(const v8h*)(ap + 32 * c);       // K = 32c + {0..7}   (+8 if hi)
        v8h a1 = *(const v8h*)(ap + 32 * c + 16);  // K = 32c + {16..23} (+8 if hi)
        af[c] = __builtin_shufflevector(a0, a1, 0, 1, 2, 3, 4, 5, 6, 7,
                                                8, 9, 10, 11, 12, 13, 14, 15);
        v8h b0 = *(const v8h*)(bp + 32 * c);       // col laneM, K = 32c + {0..7}   (+8 if hi)
        v8h b1 = *(const v8h*)(bp + 32 * c + 16);  // col laneM, K = 32c + {16..23} (+8 if hi)
        bf[c] = __builtin_shufflevector(b0, b1, 0, 1, 2, 3, 4, 5, 6, 7,
                                                8, 9, 10, 11, 12, 13, 14, 15);
    }

    v8f acc = {};
#pragma unroll
    for (int c = 0; c < NCHUNK; ++c)
        acc = __builtin_amdgcn_wmma_f32_16x16x32_f16(
            false, af[c], false, bf[c], (short)0, acc, false, false);

    // ---- Phase 3: phi -> complex rotation, coalesced b64 stores ----
    const bool fullblk = (blk + BT <= NOUT);   // uniform per block: only last block partial

#pragma unroll
    for (int v = 0; v < 8; ++v) {
        const int m   = v + hi * 8;            // C row
        const int lt  = tl + m * 8 + jcol;     // block-local output time
        const long tg = (long)blk + lt;        // global output time
        if (fullblk || tg < NOUT) {
            const float xre = XRl[(lt + N0) * 2 + omode];
            const float xim = XIl[(lt + N0) * 2 + omode];
            const float th  = (acc[v] + bmode) * coef;
            float s, c;
            __sincosf(th, &s, &c);
            float2 y;
            y.x = xre * c + xim * s;           // Re{x * exp(-i th)}
            y.y = xim * c - xre * s;           // Im{x * exp(-i th)}
            ((float2*)out)[tg * 2 + omode] = y;
        }
    }
}

extern "C" void kernel_launch(void* const* d_in, const int* in_sizes, int n_in,
                              void* d_out, int out_size, void* d_ws, size_t ws_size,
                              hipStream_t stream) {
    const float* xr    = (const float*)d_in[0];
    const float* xi    = (const float*)d_in[1];
    const float* W     = (const float*)d_in[2];
    const float* bias  = (const float*)d_in[3];
    const float* power = (const float*)d_in[4];
    float*       out   = (float*)d_out;

    const int grid = (NOUT + BT - 1) / BT;   // 4096 blocks
    nl_phase_wmma_kernel<<<grid, 256, 0, stream>>>(xr, xi, W, bias, power, out);
}
